// CellLayer_25752623907073
// MI455X (gfx1250) — matrix operand order repaired and checked by reading, not compiled, not run on hardware
//
#include <hip/hip_runtime.h>

typedef __attribute__((ext_vector_type(16))) _Float16 v16h;
typedef __attribute__((ext_vector_type(8)))  _Float16 v8h;
typedef __attribute__((ext_vector_type(2)))  _Float16 v2h;
typedef __attribute__((ext_vector_type(8)))  float    v8f;

#define B_    4
#define L_    2048
#define NIN_  32
#define NH_   64
#define H_    64
#define NS_   32
#define GD_   192            // 3*H
#define ROWS_ (B_ * L_)      // 8192
#define SC_   32             // S4D chunks
#define SCL_  64             // S4D chunk length
#define NC_   64             // DEER chunks
#define CS_   32             // DEER chunk length

// ---------------------------------------------------------------------------
// WMMA fragment layouts used here (contiguous k order, not the ISA's table order):
// A (16x32 f16): m=lane&15, K(e)=e+16*(lane>=16).
//   -> per lane: 16 contiguous K at [koff, koff+16)
// B (32x16 f16): n=lane&15, K(e)=e+16*(lane>=16).  -> 16 contiguous K per lane.
// C/D (16x16 f32): element r: m=r+8*(lane>=16), n=lane&15.
// ---------------------------------------------------------------------------
__device__ __forceinline__ v16h wmma_load_a_lds(const _Float16* base, int stride,
                                                int lane) {
  // base points at row 0 / column kk of a row-major LDS matrix; stride % 8 == 0.
  const _Float16* rowp = base + (size_t)(lane & 15) * stride + ((lane >= 16) ? 16 : 0);
  const v8h* p0 = (const v8h*)rowp;
  const v8h* p1 = (const v8h*)(rowp + 8);
  v8h a0 = *p0, a1 = *p1;
  v16h a;
#pragma unroll
  for (int e = 0; e < 8; ++e) { a[e] = a0[e]; a[8 + e] = a1[e]; }
  return a;
}

__device__ __forceinline__ v16h wmma_pack16(v8h b0, v8h b1) {
  v16h b;
#pragma unroll
  for (int e = 0; e < 8; ++e) { b[e] = b0[e]; b[8 + e] = b1[e]; }
  return b;
}

// ---------------------------------------------------------------------------
// Generic WMMA GEMM: C(MxN,f32) = A(MxK,f32 -> f16) * W(N rows x K cols, f16)
// W is the unmodified (out,in) weight layout converted to f16: B[k][n] = W[n][k],
// so each B fragment is 16 contiguous halves -> two global b128 loads.
// shiftL>0: A row r reads row r-1 of A (row r%shiftL==0 -> zero row).
// block = 128 threads (4 waves), each wave owns one 16x16 output tile.
// ---------------------------------------------------------------------------
__global__ void __launch_bounds__(128)
gemm_f32f16(const float* __restrict__ A, int lda,
            const _Float16* __restrict__ W, int N, int K,
            const float* __restrict__ bias, int relu, int shiftL,
            float* __restrict__ C) {
  int lane = threadIdx.x & 31;
  int wave = threadIdx.x >> 5;
  int tile_m = blockIdx.x;
  int n0 = blockIdx.y * 64 + wave * 16;
  if (n0 >= N) return;

  int row  = tile_m * 16 + (lane & 15);
  int koff = (lane >= 16) ? 16 : 0;
  int n    = n0 + (lane & 15);
  int kb0  = (lane >= 16) ? 16 : 0;
  const _Float16* wrow = W + (size_t)n * K;

  v8f acc = {0.f, 0.f, 0.f, 0.f, 0.f, 0.f, 0.f, 0.f};

  if (shiftL > 0) {
    int l = row & (shiftL - 1);
    bool zeroA = (l == 0);
    const float* ap = A + (size_t)(zeroA ? row : (row - 1)) * lda;
    for (int kk = 0; kk < K; kk += 32) {
      v16h a;
#pragma unroll
      for (int e = 0; e < 16; ++e) {
        int k = kk + e + koff;
        float av = ap[k];
        a[e] = zeroA ? (_Float16)0.f : (_Float16)av;
      }
      const v8h* pb = (const v8h*)(wrow + kk + kb0);
      v16h bf = wmma_pack16(pb[0], pb[1]);
      acc = __builtin_amdgcn_wmma_f32_16x16x32_f16(false, a, false, bf, (short)0, acc,
                                                   false, false);
    }
  } else {
    const float* ap = A + (size_t)row * lda;
    for (int kk = 0; kk < K; kk += 32) {
      v16h a;
#pragma unroll
      for (int e = 0; e < 16; ++e) {
        int k = kk + e + koff;
        a[e] = (_Float16)ap[k];
      }
      const v8h* pb = (const v8h*)(wrow + kk + kb0);
      v16h bf = wmma_pack16(pb[0], pb[1]);
      acc = __builtin_amdgcn_wmma_f32_16x16x32_f16(false, a, false, bf, (short)0, acc,
                                                   false, false);
    }
  }

  float bv = bias ? bias[n] : 0.f;
  int rbase = tile_m * 16 + ((lane >= 16) ? 8 : 0);
#pragma unroll
  for (int r = 0; r < 8; ++r) {
    float v = acc[r] + bv;
    if (relu) v = fmaxf(v, 0.f);
    C[(size_t)(rbase + r) * N + n] = v;
  }
}

// f32 -> f16 in-place-layout convert (weights keep their (out,in) layout)
__global__ void conv_f16(const float* __restrict__ src, _Float16* __restrict__ dst,
                         int count) {
  int idx = blockIdx.x * blockDim.x + threadIdx.x;
  if (idx < count) dst[idx] = (_Float16)src[idx];
}

// ---------------------------------------------------------------------------
// S4D derived params. P layout: [lre | lim | Ctil_re | Ctil_im | l64re | l64im],
// each NH_*NS_ = 2048 floats.  lambda = exp(dt*A),  Ctil = C*(lambda-1)/A.
// ---------------------------------------------------------------------------
__global__ void s4d_params(const float* __restrict__ log_dt, const float* __restrict__ Cre,
                           const float* __restrict__ Cim, const float* __restrict__ logA,
                           const float* __restrict__ Aim, float* __restrict__ P) {
  int idx = blockIdx.x * blockDim.x + threadIdx.x;
  if (idx >= NH_ * NS_) return;
  int h = idx >> 5;
  float dt  = __expf(log_dt[h]);
  float Are = -__expf(logA[idx]);
  float Ai  = Aim[idx];
  float xr = Are * dt, xi = Ai * dt;
  float er  = __expf(xr);
  float lre = er * cosf(xi), lim = er * sinf(xi);
  float nre = lre - 1.f, nim = lim;
  float den = Are * Are + Ai * Ai;
  float qre = (nre * Are + nim * Ai) / den;
  float qim = (nim * Are - nre * Ai) / den;
  float ctr = Cre[idx] * qre - Cim[idx] * qim;
  float cti = Cre[idx] * qim + Cim[idx] * qre;
  float e64 = __expf(64.f * xr);
  float c64 = cosf(64.f * xi), s64 = sinf(64.f * xi);
  P[idx]            = lre;
  P[2048 + idx]     = lim;
  P[2 * 2048 + idx] = ctr;
  P[3 * 2048 + idx] = cti;
  P[4 * 2048 + idx] = e64 * c64;
  P[5 * 2048 + idx] = e64 * s64;
}

// S4D phase 1: per-(b,chunk,h,mode) end-state of s = lambda*s + u over 64 steps.
__global__ void s4d_chunk_states(const float* __restrict__ U, const float* __restrict__ P,
                                 float* __restrict__ sre_o, float* __restrict__ sim_o) {
  int idx = blockIdx.x * blockDim.x + threadIdx.x;   // ((b*SC_+c)*64+h)*32+n
  int n = idx & 31;
  int h = (idx >> 5) & 63;
  int c = (idx >> 11) & 31;
  int b = idx >> 16;
  int hn = h * 32 + n;
  float lre = P[hn], lim = P[2048 + hn];
  float sr = 0.f, si = 0.f;
  const float* up = U + ((size_t)(b * L_ + c * SCL_)) * NH_ + h;
  for (int t = 0; t < SCL_; ++t) {
    float u  = up[(size_t)t * NH_];
    float nr = lre * sr - lim * si + u;
    float ni = lre * si + lim * sr;
    sr = nr; si = ni;
  }
  sre_o[idx] = sr;
  sim_o[idx] = si;
}

// S4D phase 2: carry_{c+1} = lambda^64 * carry_c + s_end_c (serial over 32 chunks).
__global__ void s4d_carry_scan(const float* __restrict__ P, const float* __restrict__ sre,
                               const float* __restrict__ sim, float* __restrict__ cre_o,
                               float* __restrict__ cim_o) {
  int idx = blockIdx.x * blockDim.x + threadIdx.x;   // (b*64+h)*32+n
  if (idx >= B_ * NH_ * NS_) return;
  int n = idx & 31, h = (idx >> 5) & 63, b = idx >> 11;
  int hn = h * 32 + n;
  float l64r = P[4 * 2048 + hn], l64i = P[5 * 2048 + hn];
  float cr = 0.f, ci = 0.f;
  for (int c = 0; c < SC_; ++c) {
    size_t e = (size_t)(((b * SC_ + c) * 64 + h) * 32 + n);
    cre_o[e] = cr;
    cim_o[e] = ci;
    float nr = l64r * cr - l64i * ci + sre[e];
    float ni = l64r * ci + l64i * cr + sim[e];
    cr = nr; ci = ni;
  }
}

// S4D phase 3: one wave per (b,h,chunk); lane = mode; cross-lane reduce per step.
__global__ void s4d_emit(const float* __restrict__ U, const float* __restrict__ P,
                         const float* __restrict__ Dv, const float* __restrict__ cre,
                         const float* __restrict__ cim, float* __restrict__ Yo) {
  int gw   = (blockIdx.x * blockDim.x + threadIdx.x) >> 5;  // b*2048 + h*32 + c
  int lane = threadIdx.x & 31;
  int c = gw & 31, h = (gw >> 5) & 63, b = gw >> 11;
  int hn = h * 32 + lane;
  float lre = P[hn], lim = P[2048 + hn];
  float ctr = P[2 * 2048 + hn], cti = P[3 * 2048 + hn];
  size_t e = (size_t)(((b * SC_ + c) * 64 + h) * 32 + lane);
  float sr = cre[e], si = cim[e];
  float dcoef = Dv[h];
  const float* up = U + ((size_t)(b * L_ + c * SCL_)) * NH_ + h;
  float* yp       = Yo + ((size_t)(b * L_ + c * SCL_)) * NH_ + h;
  for (int t = 0; t < SCL_; ++t) {
    float u  = up[(size_t)t * NH_];
    float nr = lre * sr - lim * si + u;
    float ni = lre * si + lim * sr;
    sr = nr; si = ni;
    float p = 2.f * (ctr * sr - cti * si);
#pragma unroll
    for (int off = 16; off > 0; off >>= 1) p += __shfl_xor(p, off, 32);
    if (lane == 0) yp[(size_t)t * NH_] = p + dcoef * u;
  }
}

// ---------------------------------------------------------------------------
// GRU gates + Jacobian coefficients + affine offset b_t.
// J = diag(cz) + diag(chz)*Whz + diag(chr)*Whr + diag(cha)*Wha
// b_t = f(y_prev) - J*y_prev  (uses hr/hz/ha already in HG)
// ---------------------------------------------------------------------------
__global__ void gates_kernel(const float* __restrict__ IG, const float* __restrict__ HG,
                             const float* __restrict__ Y, const float* __restrict__ bn,
                             float* __restrict__ bt, _Float16* __restrict__ cz,
                             _Float16* __restrict__ chz, _Float16* __restrict__ chr_,
                             _Float16* __restrict__ cha) {
  int idx = blockIdx.x * blockDim.x + threadIdx.x;
  if (idx >= ROWS_ * H_) return;
  int row = idx >> 6;
  int i   = idx & 63;
  int l   = row & (L_ - 1);
  float yp = (l == 0) ? 0.f : Y[(size_t)(row - 1) * H_ + i];
  const float* igr = IG + (size_t)row * GD_;
  const float* hgr = HG + (size_t)row * GD_;
  float ir = igr[i], iz = igr[64 + i], ia = igr[128 + i];
  float hr = hgr[i], hz = hgr[64 + i], ha = hgr[128 + i];
  float r  = 1.f / (1.f + __expf(-(ir + hr)));
  float z  = 1.f / (1.f + __expf(-(iz + hz)));
  float hb = ha + bn[i];
  float a  = tanhf(ia + r * hb);
  float fy = a + z * (yp - a);
  float czv = 0.f, chzv = 0.f, chrv = 0.f, chav = 0.f, btv = fy;
  if (l != 0) {
    float dz = z * (1.f - z);
    float dr = r * (1.f - r);
    float da = 1.f - a * a;
    czv  = z;
    chzv = (yp - a) * dz;
    chrv = (1.f - z) * da * hb * dr;
    chav = (1.f - z) * da * r;
    btv  = fy - (z * yp + chzv * hz + chrv * hr + chav * ha);
  }
  bt[idx]   = btv;
  cz[idx]   = (_Float16)czv;
  chz[idx]  = (_Float16)chzv;
  chr_[idx] = (_Float16)chrv;
  cha[idx]  = (_Float16)chav;
}

// ---------------------------------------------------------------------------
// DEER phase 1: per (b,chunk) sequential composition M <- J_t * M, v <- J_t*v + b_t.
// J rebuilt in LDS each step; M kept TRANSPOSED in LDS (MT[n][k]) so that
//  - B fragments are two contiguous ds_load_b128 per tile
//  - D tiles store as one contiguous ds_store_b128 per tile
// block = 128 threads = 4 waves; wave w owns output rows [16w,16w+16).
// ---------------------------------------------------------------------------
__global__ void __launch_bounds__(128)
deer_compose(const _Float16* __restrict__ whh16 /*192x64 row-major f16*/,
             const _Float16* __restrict__ cz, const _Float16* __restrict__ chz,
             const _Float16* __restrict__ chr_, const _Float16* __restrict__ cha,
             const float* __restrict__ bt, _Float16* __restrict__ Mc,
             float* __restrict__ vc) {
  __shared__ _Float16 Wr[64][66], Wz[64][66], Wa[64][66];
  __shared__ _Float16 Jm[64][72];       // row-major, stride mult of 8 for b128 loads
  __shared__ _Float16 MT[2][64][72];    // transposed M: MT[n][k] = M[k][n]
  __shared__ float    vv[2][64];

  int tid  = threadIdx.x;
  int lane = tid & 31, wave = tid >> 5;
  int blk  = blockIdx.x;
  int b = blk / NC_, c = blk - b * NC_;

  for (int idx = tid; idx < 64 * 64; idx += 128) {
    int i = idx >> 6, j = idx & 63;
    Wr[i][j]     = whh16[(size_t)i * 64 + j];
    Wz[i][j]     = whh16[(size_t)(64 + i) * 64 + j];
    Wa[i][j]     = whh16[(size_t)(128 + i) * 64 + j];
    MT[0][i][j]  = (i == j) ? (_Float16)1.f : (_Float16)0.f;   // identity (symmetric)
  }
  if (tid < 64) vv[0][tid] = 0.f;
  __syncthreads();

  int cur   = 0;
  int rbase = b * L_ + c * CS_;
  for (int t = 0; t < CS_; ++t) {
    size_t ro = (size_t)(rbase + t) * H_;
    if (t + 1 < CS_) __builtin_prefetch(&cz[ro + H_], 0, 1);
    {  // build J into LDS (2 threads per row, paired stores)
      int i  = tid >> 1;
      int j0 = (tid & 1) * 32;
      float czv  = (float)cz[ro + i];
      float chzv = (float)chz[ro + i];
      float chrv = (float)chr_[ro + i];
      float chav = (float)cha[ro + i];
#pragma unroll 4
      for (int j = j0; j < j0 + 32; j += 2) {
        v2h wz = *(const v2h*)&Wz[i][j];
        v2h wr = *(const v2h*)&Wr[i][j];
        v2h wa = *(const v2h*)&Wa[i][j];
        float w0 = chzv * (float)wz[0] + chrv * (float)wr[0] + chav * (float)wa[0];
        float w1 = chzv * (float)wz[1] + chrv * (float)wr[1] + chav * (float)wa[1];
        if (j == i) w0 += czv;
        if (j + 1 == i) w1 += czv;
        v2h pk; pk[0] = (_Float16)w0; pk[1] = (_Float16)w1;
        *(v2h*)&Jm[i][j] = pk;
      }
    }
    __syncthreads();

    // affine vector update (waves 0-1)
    float vn = 0.f;
    if (tid < 64) {
#pragma unroll 8
      for (int j = 0; j < 64; ++j) vn += (float)Jm[tid][j] * vv[cur][j];
      vn += bt[ro + tid];
    }

    // M_new = J * M_cur : 4 N-tiles x 2 K-chunks = 8 WMMAs per wave
    v8f acc[4];
#pragma unroll
    for (int n = 0; n < 4; ++n) {
      v8f z = {0.f, 0.f, 0.f, 0.f, 0.f, 0.f, 0.f, 0.f};
      acc[n] = z;
    }
#pragma unroll
    for (int kk = 0; kk < 64; kk += 32) {
      v16h a = wmma_load_a_lds(&Jm[wave * 16][kk], 72, lane);
      int kfrag = kk + ((lane >= 16) ? 16 : 0);
#pragma unroll
      for (int n = 0; n < 4; ++n) {
        int ncol = n * 16 + (lane & 15);
        const v8h* pb = (const v8h*)(&MT[cur][ncol][kfrag]);
        v16h bf = wmma_pack16(pb[0], pb[1]);
        acc[n] = __builtin_amdgcn_wmma_f32_16x16x32_f16(false, a, false, bf, (short)0,
                                                        acc[n], false, false);
      }
    }

    int nxt  = cur ^ 1;
    int mrow = wave * 16 + ((lane >= 16) ? 8 : 0);
#pragma unroll
    for (int n = 0; n < 4; ++n) {
      int ncol = n * 16 + (lane & 15);
      v8h d;
#pragma unroll
      for (int r = 0; r < 8; ++r) d[r] = (_Float16)acc[n][r];
      *(v8h*)(&MT[nxt][ncol][mrow]) = d;
    }
    if (tid < 64) vv[nxt][tid] = vn;
    __syncthreads();
    cur = nxt;
  }

  // write M (row-major) and v to global
  size_t obase = (size_t)blk * 4096;
  for (int idx = tid; idx < 4096; idx += 128)
    Mc[obase + idx] = MT[cur][idx & 63][idx >> 6];   // Mc[i*64+j] = MT[j][i]
  if (tid < 64) vc[(size_t)blk * 64 + tid] = vv[cur][tid];
}

// DEER phase 2: serial carry across chunks. One block (64 thr) per batch.
__global__ void deer_scan(const _Float16* __restrict__ Mc, const float* __restrict__ vc,
                          float* __restrict__ carry) {
  int b = blockIdx.x, tid = threadIdx.x;
  __shared__ float hs[64];
  float h = 0.f;
  for (int c = 0; c < NC_; ++c) {
    size_t base = (size_t)(b * NC_ + c);
    carry[base * 64 + tid] = h;   // state entering chunk c
    hs[tid] = h;
    __syncthreads();
    float acc = vc[base * 64 + tid];
    const _Float16* mrow = Mc + base * 4096 + (size_t)tid * 64;
#pragma unroll 8
    for (int j = 0; j < 64; ++j) acc += (float)mrow[j] * hs[j];
    __syncthreads();
    h = acc;
  }
}

// DEER phase 3: re-walk chunk with carry-in, matvecs only. 64 threads/block.
__global__ void __launch_bounds__(64)
deer_apply(const float* __restrict__ w_hh, const _Float16* __restrict__ cz,
           const _Float16* __restrict__ chz, const _Float16* __restrict__ chr_,
           const _Float16* __restrict__ cha, const float* __restrict__ bt,
           const float* __restrict__ carry, float* __restrict__ yout) {
  __shared__ float Wr[64][65], Wz[64][65], Wa[64][65];
  __shared__ float ys[64];
  int tid = threadIdx.x;
  int blk = blockIdx.x;
  int b = blk / NC_, c = blk - b * NC_;
  for (int idx = tid; idx < 4096; idx += 64) {
    int i = idx >> 6, j = idx & 63;
    Wr[i][j] = w_hh[(size_t)i * 64 + j];
    Wz[i][j] = w_hh[(size_t)(64 + i) * 64 + j];
    Wa[i][j] = w_hh[(size_t)(128 + i) * 64 + j];
  }
  ys[tid] = carry[(size_t)blk * 64 + tid];
  __syncthreads();
  int rbase = b * L_ + c * CS_;
  for (int t = 0; t < CS_; ++t) {
    size_t ro = (size_t)(rbase + t) * H_;
    float s1 = 0.f, s2 = 0.f, s3 = 0.f;
#pragma unroll 8
    for (int j = 0; j < 64; ++j) {
      float y = ys[j];
      s1 += Wr[tid][j] * y;
      s2 += Wz[tid][j] * y;
      s3 += Wa[tid][j] * y;
    }
    float yn = (float)cz[ro + tid] * ys[tid] + (float)chz[ro + tid] * s2 +
               (float)chr_[ro + tid] * s1 + (float)cha[ro + tid] * s3 + bt[ro + tid];
    __syncthreads();
    ys[tid] = yn;
    yout[ro + tid] = yn;
    __syncthreads();
  }
}

// ---------------------------------------------------------------------------
// Host driver
// ---------------------------------------------------------------------------
extern "C" void kernel_launch(void* const* d_in, const int* in_sizes, int n_in,
                              void* d_out, int out_size, void* d_ws, size_t ws_size,
                              hipStream_t stream) {
  (void)in_sizes; (void)n_in; (void)out_size; (void)ws_size;
  const float* xs        = (const float*)d_in[0];
  const float* m1w1      = (const float*)d_in[1];
  const float* m1b1      = (const float*)d_in[2];
  const float* m1w2      = (const float*)d_in[3];
  const float* m1b2      = (const float*)d_in[4];
  const float* s1_log_dt = (const float*)d_in[5];
  const float* s1_Cre    = (const float*)d_in[6];
  const float* s1_Cim    = (const float*)d_in[7];
  const float* s1_logA   = (const float*)d_in[8];
  const float* s1_Aim    = (const float*)d_in[9];
  const float* s1_D      = (const float*)d_in[10];
  const float* m2w1      = (const float*)d_in[11];
  const float* m2b1      = (const float*)d_in[12];
  const float* m2w2      = (const float*)d_in[13];
  const float* m2b2      = (const float*)d_in[14];
  const float* s2_log_dt = (const float*)d_in[15];
  const float* s2_Cre    = (const float*)d_in[16];
  const float* s2_Cim    = (const float*)d_in[17];
  const float* s2_logA   = (const float*)d_in[18];
  const float* s2_Aim    = (const float*)d_in[19];
  const float* s2_D      = (const float*)d_in[20];
  const float* m3w1      = (const float*)d_in[21];
  const float* m3b1      = (const float*)d_in[22];
  const float* m3w2      = (const float*)d_in[23];
  const float* m3b2      = (const float*)d_in[24];
  const float* w_ih      = (const float*)d_in[25];
  const float* w_hh      = (const float*)d_in[26];
  const float* b_gru     = (const float*)d_in[27];
  const float* bn_gru    = (const float*)d_in[28];

  char* ws = (char*)d_ws;
  size_t off = 0;
  auto alloc = [&](size_t bytes) -> char* {
    char* p = ws + off;
    off = (off + bytes + 255) & ~(size_t)255;
    return p;
  };

  _Float16* w1f  = (_Float16*)alloc(64 * 32 * 2);
  _Float16* w2f  = (_Float16*)alloc(64 * 64 * 2);
  _Float16* w3f  = (_Float16*)alloc(64 * 64 * 2);
  _Float16* w4f  = (_Float16*)alloc(64 * 64 * 2);
  _Float16* w5f  = (_Float16*)alloc(64 * 64 * 2);
  _Float16* w6f  = (_Float16*)alloc(64 * 64 * 2);
  _Float16* wihf = (_Float16*)alloc(192 * 32 * 2);
  _Float16* whhf = (_Float16*)alloc(192 * 64 * 2);
  float* P1 = (float*)alloc(6 * 2048 * 4);
  float* P2 = (float*)alloc(6 * 2048 * 4);
  float* ACT0 = (float*)alloc((size_t)ROWS_ * 64 * 4);
  float* ACT1 = (float*)alloc((size_t)ROWS_ * 64 * 4);
  float* ACT2 = (float*)alloc((size_t)ROWS_ * 64 * 4);
  float* YA   = (float*)alloc((size_t)ROWS_ * 64 * 4);
  float* YB   = (float*)alloc((size_t)ROWS_ * 64 * 4);
  float* IG   = (float*)alloc((size_t)ROWS_ * 192 * 4);
  float* HG   = (float*)alloc((size_t)ROWS_ * 192 * 4);
  _Float16* cz  = (_Float16*)alloc((size_t)ROWS_ * 64 * 2);
  _Float16* chz = (_Float16*)alloc((size_t)ROWS_ * 64 * 2);
  _Float16* chr = (_Float16*)alloc((size_t)ROWS_ * 64 * 2);
  _Float16* cha = (_Float16*)alloc((size_t)ROWS_ * 64 * 2);
  float* bt  = (float*)alloc((size_t)ROWS_ * 64 * 4);
  float* sre = (float*)alloc((size_t)B_ * SC_ * 64 * 32 * 4);
  float* sim = (float*)alloc((size_t)B_ * SC_ * 64 * 32 * 4);
  float* scr = (float*)alloc((size_t)B_ * SC_ * 64 * 32 * 4);
  float* sci = (float*)alloc((size_t)B_ * SC_ * 64 * 32 * 4);
  _Float16* Mc = (_Float16*)alloc((size_t)B_ * NC_ * 4096 * 2);
  float* vc   = (float*)alloc((size_t)B_ * NC_ * 64 * 4);
  float* hcar = (float*)alloc((size_t)B_ * NC_ * 64 * 4);

  // --- weight prep (f32 -> f16, layout preserved) ---
  conv_f16<<<(64 * 32 + 255) / 256, 256, 0, stream>>>(m1w1, w1f, 64 * 32);
  conv_f16<<<(64 * 64 + 255) / 256, 256, 0, stream>>>(m1w2, w2f, 64 * 64);
  conv_f16<<<(64 * 64 + 255) / 256, 256, 0, stream>>>(m2w1, w3f, 64 * 64);
  conv_f16<<<(64 * 64 + 255) / 256, 256, 0, stream>>>(m2w2, w4f, 64 * 64);
  conv_f16<<<(64 * 64 + 255) / 256, 256, 0, stream>>>(m3w1, w5f, 64 * 64);
  conv_f16<<<(64 * 64 + 255) / 256, 256, 0, stream>>>(m3w2, w6f, 64 * 64);
  conv_f16<<<(192 * 32 + 255) / 256, 256, 0, stream>>>(w_ih, wihf, 192 * 32);
  conv_f16<<<(192 * 64 + 255) / 256, 256, 0, stream>>>(w_hh, whhf, 192 * 64);
  s4d_params<<<8, 256, 0, stream>>>(s1_log_dt, s1_Cre, s1_Cim, s1_logA, s1_Aim, P1);
  s4d_params<<<8, 256, 0, stream>>>(s2_log_dt, s2_Cre, s2_Cim, s2_logA, s2_Aim, P2);

  dim3 g1(ROWS_ / 16, 1), g3(ROWS_ / 16, 3), b128(128);

  // --- init-guess network: MLP -> S4D -> MLP -> S4D -> MLP ---
  gemm_f32f16<<<g1, b128, 0, stream>>>(xs, NIN_, w1f, 64, 32, m1b1, 1, 0, ACT0);
  gemm_f32f16<<<g1, b128, 0, stream>>>(ACT0, 64, w2f, 64, 64, m1b2, 0, 0, ACT1);
  s4d_chunk_states<<<1024, 256, 0, stream>>>(ACT1, P1, sre, sim);
  s4d_carry_scan<<<32, 256, 0, stream>>>(P1, sre, sim, scr, sci);
  s4d_emit<<<1024, 256, 0, stream>>>(ACT1, P1, s1_D, scr, sci, ACT2);
  gemm_f32f16<<<g1, b128, 0, stream>>>(ACT2, 64, w3f, 64, 64, m2b1, 1, 0, ACT0);
  gemm_f32f16<<<g1, b128, 0, stream>>>(ACT0, 64, w4f, 64, 64, m2b2, 0, 0, ACT1);
  s4d_chunk_states<<<1024, 256, 0, stream>>>(ACT1, P2, sre, sim);
  s4d_carry_scan<<<32, 256, 0, stream>>>(P2, sre, sim, scr, sci);
  s4d_emit<<<1024, 256, 0, stream>>>(ACT1, P2, s2_D, scr, sci, ACT2);
  gemm_f32f16<<<g1, b128, 0, stream>>>(ACT2, 64, w5f, 64, 64, m3b1, 1, 0, ACT0);
  gemm_f32f16<<<g1, b128, 0, stream>>>(ACT0, 64, w6f, 64, 64, m3b2, 0, 0, YA);

  // --- DEER quasi-Newton iterations ---
  gemm_f32f16<<<g3, b128, 0, stream>>>(xs, NIN_, wihf, 192, 32, b_gru, 0, 0, IG);
  float* yin = YA;
  for (int it = 0; it < 5; ++it) {
    gemm_f32f16<<<g3, b128, 0, stream>>>(yin, 64, whhf, 192, 64, nullptr, 0, L_, HG);
    gates_kernel<<<(ROWS_ * H_) / 256, 256, 0, stream>>>(IG, HG, yin, bn_gru, bt, cz,
                                                         chz, chr, cha);
    deer_compose<<<B_ * NC_, 128, 0, stream>>>(whhf, cz, chz, chr, cha, bt, Mc, vc);
    deer_scan<<<B_, 64, 0, stream>>>(Mc, vc, hcar);
    float* yout = (it == 4) ? (float*)d_out : ((it & 1) ? YA : YB);
    deer_apply<<<B_ * NC_, 64, 0, stream>>>(w_hh, cz, chz, chr, cha, bt, hcar, yout);
    yin = yout;
  }
}
